// CausalSelfAttention_41180146434492
// MI455X (gfx1250) — matrix operand order repaired and checked by reading, not compiled, not run on hardware
//
#include <hip/hip_runtime.h>

// ---------------------------------------------------------------------------
// Types matching gfx1250 WMMA operand shapes
// ---------------------------------------------------------------------------
typedef __bf16 bf16;
typedef __attribute__((ext_vector_type(8)))  __bf16 v8bf;
typedef __attribute__((ext_vector_type(16))) __bf16 v16bf;
typedef __attribute__((ext_vector_type(8)))  float  v8f;

// Problem constants (reference: B=4, S=2048, D=1024, H=16, HD=64)
constexpr int BB  = 4;
constexpr int SS  = 2048;
constexpr int DD  = 1024;
constexpr int HH  = 16;
constexpr int HD  = 64;
constexpr int MM  = BB * SS;      // 8192 rows
constexpr int N3  = 3 * DD;       // 3072

__device__ __forceinline__ v8bf ld8(const bf16* p) {
    return *reinterpret_cast<const v8bf*>(p);
}
__device__ __forceinline__ v16bf cat16(v8bf lo, v8bf hi) {
    return __builtin_shufflevector(lo, hi, 0,1,2,3,4,5,6,7,8,9,10,11,12,13,14,15);
}
__device__ __forceinline__ v8f wmma_bf16(v16bf a, v16bf b, v8f c) {
    return __builtin_amdgcn_wmma_f32_16x16x32_bf16(
        false, a, false, b, (short)0, c, false, false);
}
__device__ __forceinline__ float rowmax16(float v) {
    #pragma unroll
    for (int m = 1; m < 16; m <<= 1) v = fmaxf(v, __shfl_xor(v, m, 32));
    return v;
}
__device__ __forceinline__ float rowsum16(float v) {
    #pragma unroll
    for (int m = 1; m < 16; m <<= 1) v += __shfl_xor(v, m, 32);
    return v;
}

// ---------------------------------------------------------------------------
// Kernel 1: fp32 -> bf16 elementwise convert
// ---------------------------------------------------------------------------
__global__ void csa_cvt_bf16(const float* __restrict__ src,
                             bf16* __restrict__ dst, size_t n) {
    size_t i = (size_t)blockIdx.x * blockDim.x + threadIdx.x;
    size_t stride = (size_t)gridDim.x * blockDim.x;
    for (; i < n; i += stride) dst[i] = (bf16)src[i];
}

// ---------------------------------------------------------------------------
// Kernel 2: fp32 [K,N] -> bf16 [N,K] transpose+convert (weights)
// ---------------------------------------------------------------------------
__global__ void csa_transpose_bf16(const float* __restrict__ src,
                                   bf16* __restrict__ dst, int K, int N) {
    size_t total = (size_t)K * N;
    size_t i = (size_t)blockIdx.x * blockDim.x + threadIdx.x;
    size_t stride = (size_t)gridDim.x * blockDim.x;
    for (; i < total; i += stride) {
        int k = (int)(i / N);
        int n = (int)(i - (size_t)k * N);
        dst[(size_t)n * K + k] = (bf16)src[i];
    }
}

// ---------------------------------------------------------------------------
// GEMM fragment bundle: one k-step of loads for a 32(M)x64(N) wave tile
// ---------------------------------------------------------------------------
struct Frags {
    v8bf a[2][2];   // [mtile][lo/hi]  A-layout: lane row, K split per half-wave
    v8bf b[4][2];   // [ntile][lo/hi]  B-layout: lane col, 16 contiguous K
};

__device__ __forceinline__ void load_frags(Frags& f,
                                           const bf16* xrow0,
                                           const bf16* xrow1,
                                           const bf16* const wrow[4],
                                           int kk, int hi) {
    f.a[0][0] = ld8(xrow0 + kk + hi * 16);
    f.a[0][1] = ld8(xrow0 + kk + hi * 16 + 8);
    f.a[1][0] = ld8(xrow1 + kk + hi * 16);
    f.a[1][1] = ld8(xrow1 + kk + hi * 16 + 8);
    #pragma unroll
    for (int s = 0; s < 4; ++s) {
        f.b[s][0] = ld8(wrow[s] + kk);
        f.b[s][1] = ld8(wrow[s] + kk + 8);
    }
}

__device__ __forceinline__ void do_wmmas(v8f acc[2][4], const Frags& f) {
    v16bf a0 = cat16(f.a[0][0], f.a[0][1]);
    v16bf a1 = cat16(f.a[1][0], f.a[1][1]);
    #pragma unroll
    for (int s = 0; s < 4; ++s) {
        v16bf b = cat16(f.b[s][0], f.b[s][1]);
        acc[0][s] = wmma_bf16(a0, b, acc[0][s]);
        acc[1][s] = wmma_bf16(a1, b, acc[1][s]);
    }
}

// ---------------------------------------------------------------------------
// Kernel 3: QKV GEMM  [8192,1024]x[1024,3072] + bias, bf16 WMMA, pipelined.
// Wave tile 32x64, block = 8 waves = 256 rows x 64 cols.
// Scatter Q,K -> [B,H,S,HD]; V -> [B,H,HD,S]
// ---------------------------------------------------------------------------
__global__ __launch_bounds__(256)
void csa_qkv_gemm(const bf16* __restrict__ X,
                  const bf16* __restrict__ Wt,   // [3072,1024] N-major
                  const float* __restrict__ bias,
                  bf16* __restrict__ Qb, bf16* __restrict__ Kb,
                  bf16* __restrict__ Vt) {
    const int wave = threadIdx.x >> 5;
    const int lane = threadIdx.x & 31;
    const int ln   = lane & 15;
    const int hi   = lane >> 4;
    const int m0   = blockIdx.x * 256 + wave * 32;
    const int nb   = blockIdx.y * 64;

    const bf16* xrow0 = X + (size_t)(m0 + ln) * DD;
    const bf16* xrow1 = X + (size_t)(m0 + 16 + ln) * DD;
    const bf16* wrow[4];
    #pragma unroll
    for (int s = 0; s < 4; ++s)
        wrow[s] = Wt + (size_t)(nb + s * 16 + ln) * DD + hi * 16;

    v8f acc[2][4] = {};
    Frags cur, nxt;
    load_frags(cur, xrow0, xrow1, wrow, 0, hi);
    for (int kk = 32; kk < DD; kk += 32) {
        load_frags(nxt, xrow0, xrow1, wrow, kk, hi);  // in flight during WMMAs
        do_wmmas(acc, cur);
        cur = nxt;
    }
    do_wmmas(acc, cur);

    #pragma unroll
    for (int s = 0; s < 4; ++s) {
        const int n  = nb + s * 16 + ln;
        const float bv = bias[n];
        const int which = n >> 10;        // 0=Q 1=K 2=V
        const int d  = n & (DD - 1);
        const int h  = d >> 6;
        const int hd = d & (HD - 1);
        #pragma unroll
        for (int t = 0; t < 2; ++t) {
            #pragma unroll
            for (int r = 0; r < 8; ++r) {
                const int m = m0 + t * 16 + r + hi * 8;
                const int b = m >> 11;
                const int sIdx = m & (SS - 1);
                const int bh = b * HH + h;
                const bf16 val = (bf16)(acc[t][s][r] + bv);
                if (which == 0)
                    Qb[((size_t)bh * SS + sIdx) * HD + hd] = val;
                else if (which == 1)
                    Kb[((size_t)bh * SS + sIdx) * HD + hd] = val;
                else
                    Vt[((size_t)bh * HD + hd) * SS + sIdx] = val;
            }
        }
    }
}

// ---------------------------------------------------------------------------
// Kernel 4: flash attention. One wave per 16-query tile.
// V loads issued before softmax VALU block to hide latency.
// ---------------------------------------------------------------------------
__global__ __launch_bounds__(128)
void csa_attention(const bf16* __restrict__ Qb,
                   const bf16* __restrict__ Kb,
                   const bf16* __restrict__ Vt,   // [B*H, HD, S]
                   bf16* __restrict__ Ctx) {      // [B*S, D] bf16
    __shared__ __align__(64) bf16 plds[4][16 * 32];

    const int wave = threadIdx.x >> 5;
    const int lane = threadIdx.x & 31;
    const int ln   = lane & 15;
    const int hi   = lane >> 4;

    const int tile = blockIdx.x * 4 + wave;
    const int bh   = tile >> 7;
    const int qb   = (tile & 127) * 16;

    const bf16* qrow = Qb + ((size_t)bh * SS + qb + ln) * HD;
    const v16bf aq0 = cat16(ld8(qrow + hi * 16),      ld8(qrow + hi * 16 + 8));
    const v16bf aq1 = cat16(ld8(qrow + 32 + hi * 16), ld8(qrow + 40 + hi * 16));

    float mrow[8], lrow[8];
    v8f acc[4] = {};
    #pragma unroll
    for (int r = 0; r < 8; ++r) { mrow[r] = -3.0e38f; lrow[r] = 0.0f; }

    const bf16* kpage = Kb + (size_t)bh * SS * HD;
    const bf16* vpage = Vt + (size_t)bh * HD * SS;
    bf16* pl = plds[wave];

    const int nblk = (qb + 47) >> 5;
    for (int kb = 0; kb < nblk; ++kb) {
        const int k0 = kb * 32;

        // ---- issue K loads, then V loads (V latency hidden by softmax) ----
        const bf16* kr0 = kpage + (size_t)(k0 + ln) * HD + hi * 16;
        const bf16* kr1 = kpage + (size_t)(k0 + 16 + ln) * HD + hi * 16;
        v8bf k00 = ld8(kr0), k01 = ld8(kr0 + 8);
        v8bf k02 = ld8(kr0 + 32), k03 = ld8(kr0 + 40);
        v8bf k10 = ld8(kr1), k11 = ld8(kr1 + 8);
        v8bf k12 = ld8(kr1 + 32), k13 = ld8(kr1 + 40);

        v8bf vlo[4], vhi[4];
        #pragma unroll
        for (int s = 0; s < 4; ++s) {
            const bf16* vr = vpage + (size_t)(s * 16 + ln) * SS + k0 + hi * 16;
            vlo[s] = ld8(vr);
            vhi[s] = ld8(vr + 8);
        }
        if (kb + 1 < nblk) {   // prefetch next key block (contiguous rows)
            __builtin_prefetch(kpage + (size_t)(k0 + 32 + ln) * HD, 0, 0);
            __builtin_prefetch(kpage + (size_t)(k0 + 48 + ln) * HD, 0, 0);
        }

        // ---- scores ----
        v8f s0 = {}, s1 = {};
        s0 = wmma_bf16(aq0, cat16(k00, k01), s0);
        s0 = wmma_bf16(aq1, cat16(k02, k03), s0);
        s1 = wmma_bf16(aq0, cat16(k10, k11), s1);
        s1 = wmma_bf16(aq1, cat16(k12, k13), s1);

        // ---- scale, causal mask, online softmax ----
        float p0[8], p1[8];
        const int n0 = k0 + ln, n1 = k0 + 16 + ln;
        #pragma unroll
        for (int r = 0; r < 8; ++r) {
            const int mg = qb + r + hi * 8;
            float a0 = s0[r] * 0.125f;
            float a1 = s1[r] * 0.125f;
            if (n0 > mg) a0 = -10000.0f;
            if (n1 > mg) a1 = -10000.0f;
            p0[r] = a0; p1[r] = a1;
        }
        #pragma unroll
        for (int r = 0; r < 8; ++r) {
            const float mx    = rowmax16(fmaxf(p0[r], p1[r]));
            const float mnew  = fmaxf(mrow[r], mx);
            const float alpha = __expf(mrow[r] - mnew);
            p0[r] = __expf(p0[r] - mnew);
            p1[r] = __expf(p1[r] - mnew);
            const float ls = rowsum16(p0[r] + p1[r]);
            lrow[r] = lrow[r] * alpha + ls;
            mrow[r] = mnew;
            #pragma unroll
            for (int s = 0; s < 4; ++s) acc[s][r] *= alpha;
        }

        // ---- P: C-layout regs -> LDS -> A-layout fragment ----
        #pragma unroll
        for (int r = 0; r < 8; ++r) {
            const int row = r + hi * 8;
            pl[row * 32 + ln]      = (bf16)p0[r];
            pl[row * 32 + 16 + ln] = (bf16)p1[r];
        }
        asm volatile("s_wait_dscnt 0x0" ::: "memory");  // wave-private tile
        const v8bf* plv = reinterpret_cast<const v8bf*>(pl + ln * 32);
        v16bf ap = cat16(plv[2 * hi], plv[2 * hi + 1]);

        // ---- O += P @ V (V already resident) ----
        #pragma unroll
        for (int s = 0; s < 4; ++s)
            acc[s] = wmma_bf16(ap, cat16(vlo[s], vhi[s]), acc[s]);
    }

    const int b = bh >> 4, h = bh & 15;
    #pragma unroll
    for (int r = 0; r < 8; ++r) {
        const float inv = 1.0f / lrow[r];
        const int token = b * SS + qb + r + hi * 8;
        #pragma unroll
        for (int s = 0; s < 4; ++s) {
            Ctx[(size_t)token * DD + h * HD + s * 16 + ln] =
                (bf16)(acc[s][r] * inv);
        }
    }
}

// ---------------------------------------------------------------------------
// Kernel 5: output projection  [8192,1024]x[1024,1024] + bias -> fp32
// ---------------------------------------------------------------------------
__global__ __launch_bounds__(256)
void csa_proj_gemm(const bf16* __restrict__ Ctx,
                   const bf16* __restrict__ Wt,   // [1024,1024] N-major
                   const float* __restrict__ bias,
                   float* __restrict__ out) {
    const int wave = threadIdx.x >> 5;
    const int lane = threadIdx.x & 31;
    const int ln   = lane & 15;
    const int hi   = lane >> 4;
    const int m0   = blockIdx.x * 256 + wave * 32;
    const int nb   = blockIdx.y * 64;

    const bf16* xrow0 = Ctx + (size_t)(m0 + ln) * DD;
    const bf16* xrow1 = Ctx + (size_t)(m0 + 16 + ln) * DD;
    const bf16* wrow[4];
    #pragma unroll
    for (int s = 0; s < 4; ++s)
        wrow[s] = Wt + (size_t)(nb + s * 16 + ln) * DD + hi * 16;

    v8f acc[2][4] = {};
    Frags cur, nxt;
    load_frags(cur, xrow0, xrow1, wrow, 0, hi);
    for (int kk = 32; kk < DD; kk += 32) {
        load_frags(nxt, xrow0, xrow1, wrow, kk, hi);
        do_wmmas(acc, cur);
        cur = nxt;
    }
    do_wmmas(acc, cur);

    #pragma unroll
    for (int s = 0; s < 4; ++s) {
        const int n = nb + s * 16 + ln;
        const float bv = bias[n];
        #pragma unroll
        for (int t = 0; t < 2; ++t) {
            #pragma unroll
            for (int r = 0; r < 8; ++r) {
                const int m = m0 + t * 16 + r + hi * 8;
                out[(size_t)m * DD + n] = acc[t][s][r] + bv;
            }
        }
    }
}

// ---------------------------------------------------------------------------
// Host launcher
// ---------------------------------------------------------------------------
extern "C" void kernel_launch(void* const* d_in, const int* in_sizes, int n_in,
                              void* d_out, int out_size, void* d_ws, size_t ws_size,
                              hipStream_t stream) {
    const float* hidden  = (const float*)d_in[0];
    const float* w_attn  = (const float*)d_in[1];
    const float* b_attn  = (const float*)d_in[2];
    const float* w_proj  = (const float*)d_in[3];
    const float* b_proj  = (const float*)d_in[4];
    float* out = (float*)d_out;

    char* ws = (char*)d_ws;
    const size_t SZ_X  = (size_t)MM * DD * sizeof(bf16);
    const size_t SZ_WA = (size_t)N3 * DD * sizeof(bf16);
    const size_t SZ_WP = (size_t)DD * DD * sizeof(bf16);
    const size_t SZ_H  = (size_t)BB * HH * SS * HD * sizeof(bf16);
    bf16* Xbf = (bf16*)(ws);
    bf16* WaT = (bf16*)(ws + SZ_X);
    bf16* WpT = (bf16*)(ws + SZ_X + SZ_WA);
    bf16* Qb  = (bf16*)(ws + SZ_X + SZ_WA + SZ_WP);
    bf16* Kb  = (bf16*)(ws + SZ_X + SZ_WA + SZ_WP + SZ_H);
    bf16* Vt  = (bf16*)(ws + SZ_X + SZ_WA + SZ_WP + 2 * SZ_H);
    bf16* Ctx = (bf16*)(ws + SZ_X + SZ_WA + SZ_WP + 3 * SZ_H);

    csa_cvt_bf16<<<2048, 256, 0, stream>>>(hidden, Xbf, (size_t)MM * DD);
    csa_transpose_bf16<<<2048, 256, 0, stream>>>(w_attn, WaT, DD, N3);
    csa_transpose_bf16<<<1024, 256, 0, stream>>>(w_proj, WpT, DD, DD);

    csa_qkv_gemm<<<dim3(MM / 256, N3 / 64), 256, 0, stream>>>(
        Xbf, WaT, b_attn, Qb, Kb, Vt);

    csa_attention<<<(BB * HH * (SS / 16)) / 4, 128, 0, stream>>>(Qb, Kb, Vt, Ctx);

    csa_proj_gemm<<<dim3(MM / 256, DD / 64), 256, 0, stream>>>(
        Ctx, WpT, b_proj, out);
}
